// Fusion_35639638622775
// MI455X (gfx1250) — matrix operand order repaired and checked by reading, not compiled, not run on hardware
//
#include <hip/hip_runtime.h>
#include <cmath>

// ---------------------------------------------------------------------------
// MI455X (gfx1250) implementation. wave32, WMMA f32_16x16x32_f16 for all
// matmul-shaped work; async Global->LDS DMA (ASYNCcnt) for operand staging.
// ---------------------------------------------------------------------------

typedef _Float16 v16h __attribute__((ext_vector_type(16)));
typedef _Float16 h8   __attribute__((ext_vector_type(8)));
typedef float    v8f  __attribute__((ext_vector_type(8)));

#define WMMA(a,b,c) __builtin_amdgcn_wmma_f32_16x16x32_f16(false,(a),false,(b),(short)0,(c),false,false)

__device__ __forceinline__ v16h cat16(h8 lo, h8 hi) {
  return __builtin_shufflevector(lo, hi, 0,1,2,3,4,5,6,7,8,9,10,11,12,13,14,15);
}

// LDS byte-offset of a __shared__ object (generic -> AS3 -> 32-bit offset)
__device__ __forceinline__ unsigned lds_off32(const void* p) {
  return (unsigned)(size_t)(const __attribute__((address_space(3))) void*)p;
}

// 16-byte Global -> LDS async DMA (GLOBAL_LOAD_ASYNC_TO_LDS_B128, ASYNCcnt)
__device__ __forceinline__ void async_copy16(const float* gp, const void* ldsp) {
  unsigned lo = lds_off32(ldsp);
  asm volatile("global_load_async_to_lds_b128 %0, %1, off" :: "v"(lo), "v"(gp) : "memory");
}

__device__ __forceinline__ void wait_async0() {
#if defined(__has_builtin)
#if __has_builtin(__builtin_amdgcn_s_wait_asynccnt)
  __builtin_amdgcn_s_wait_asynccnt(0);
#else
  asm volatile("s_wait_asynccnt 0x0" ::: "memory");
#endif
#else
  asm volatile("s_wait_asynccnt 0x0" ::: "memory");
#endif
}

// Build an A-fragment (16x32, K=hi*16+e) from an f32 LDS row
__device__ __forceinline__ v16h frag_a_f32(const float* rowp, int hi) {
  float4 a0 = *(const float4*)(rowp + hi * 16);
  float4 a1 = *(const float4*)(rowp + hi * 16 + 4);
  float4 a2 = *(const float4*)(rowp + hi * 16 + 8);
  float4 a3 = *(const float4*)(rowp + hi * 16 + 12);
  v16h f;
  f[0]=(_Float16)a0.x; f[1]=(_Float16)a0.y; f[2]=(_Float16)a0.z; f[3]=(_Float16)a0.w;
  f[4]=(_Float16)a1.x; f[5]=(_Float16)a1.y; f[6]=(_Float16)a1.z; f[7]=(_Float16)a1.w;
  f[8]=(_Float16)a2.x; f[9]=(_Float16)a2.y; f[10]=(_Float16)a2.z; f[11]=(_Float16)a2.w;
  f[12]=(_Float16)a3.x; f[13]=(_Float16)a3.y; f[14]=(_Float16)a3.z; f[15]=(_Float16)a3.w;
  return f;
}

// Build a B-fragment (32x16, K=hi*16+e) from an f32 LDS row (n-major)
__device__ __forceinline__ v16h frag_b_f32(const float* rowp, int hi) {
  float4 b0 = *(const float4*)(rowp + hi * 16);
  float4 b1 = *(const float4*)(rowp + hi * 16 + 4);
  float4 b2 = *(const float4*)(rowp + hi * 16 + 8);
  float4 b3 = *(const float4*)(rowp + hi * 16 + 12);
  v16h f;
  f[0]=(_Float16)b0.x; f[1]=(_Float16)b0.y; f[2]=(_Float16)b0.z; f[3]=(_Float16)b0.w;
  f[4]=(_Float16)b1.x; f[5]=(_Float16)b1.y; f[6]=(_Float16)b1.z; f[7]=(_Float16)b1.w;
  f[8]=(_Float16)b2.x; f[9]=(_Float16)b2.y; f[10]=(_Float16)b2.z; f[11]=(_Float16)b2.w;
  f[12]=(_Float16)b3.x; f[13]=(_Float16)b3.y; f[14]=(_Float16)b3.z; f[15]=(_Float16)b3.w;
  return f;
}

// constants
#define BB    2
#define DD    128
#define HH    64
#define WW    1024
#define PP    65536      // H*W
#define C3D_  256
#define KCONV 2304       // 256*9
#define PR    1920       // 24*80
#define NQ_   3000
#define NCLS_ 20
#define DFF_  512

// ---------------------------------------------------------------------------
// Generic WMMA GEMM:  C[M,N] = act( A[M,K] * W + bias (+resid) )
// AMODE 0: A row-major [M,K] (async DMA staged). AMODE 1: A=raw[(b*K+k)*bP+p]
// WMODE 0: W[k*N+n] (x @ W, transpose-staged). WMODE 1: W[n*K+k] (async DMA)
// Block 256 thr (8 waves), tile M=64 x N=128, K-step 32 through f32 LDS.
// ---------------------------------------------------------------------------
template<int AMODE, int WMODE>
__global__ __launch_bounds__(256)
void gemm_wmma_k(const float* __restrict__ A, const float* __restrict__ Wt,
                 const float* __restrict__ bias, const float* __restrict__ resid,
                 float* __restrict__ C, int M, int N, int K, int batchP, int relu)
{
  __shared__ float Asf[64][32];
  __shared__ float Bsf[128][32];
  const int tid = threadIdx.x, lane = tid & 31, wave = tid >> 5;
  const int m0 = blockIdx.x * 64;
  const int n0 = blockIdx.y * 128;
  const int msub = wave & 3, nhalf = wave >> 2;
  const int hi = lane >> 4, ln = lane & 15;

  v8f acc[4];
  for (int t = 0; t < 4; ++t) for (int r = 0; r < 8; ++r) acc[t][r] = 0.f;

  for (int k0 = 0; k0 < K; k0 += 32) {
    __syncthreads();
    // ---- stage A tile (64 x 32 f32) ----
    if (AMODE == 0) {
      // rows are K-contiguous: straight DMA, 2 x 16B per thread
      for (int j = 0; j < 2; ++j) {
        int chunk = j * 256 + tid;        // 0..511
        int rowl  = chunk >> 3;           // 0..63
        int koff  = (chunk & 7) << 2;     // 0..28
        int row = m0 + rowl; if (row >= M) row = M - 1;
        async_copy16(A + (size_t)row * K + k0 + koff, &Asf[rowl][koff]);
      }
    } else {
      for (int i = tid; i < 64 * 32; i += 256) {
        int m = i >> 5, k = i & 31;
        int row = m0 + m; if (row >= M) row = M - 1;
        int kk = k0 + k;
        float v = 0.f;
        if (kk < K) {
          int b = row / batchP, p = row - b * batchP;
          v = A[((size_t)(b * K) + kk) * batchP + p];
        }
        Asf[m][k] = v;
      }
    }
    // ---- stage B^T tile (128n x 32k f32) ----
    if (WMODE == 1) {
      // weight rows are K-contiguous per n: straight DMA, 4 x 16B per thread
      for (int j = 0; j < 4; ++j) {
        int chunk = j * 256 + tid;        // 0..1023
        int nl   = chunk >> 3;            // 0..127
        int koff = (chunk & 7) << 2;      // 0..28
        int nn = n0 + nl; if (nn >= N) nn = N - 1;
        async_copy16(Wt + (size_t)nn * K + k0 + koff, &Bsf[nl][koff]);
      }
    } else {
      for (int i = tid; i < 128 * 32; i += 256) {
        int n = i >> 5, k = i & 31;
        int kk = k0 + k, nn = n0 + n;
        Bsf[n][k] = (kk < K && nn < N) ? Wt[(size_t)kk * N + nn] : 0.f;
      }
    }
    if (k0 + 32 < K)  // prefetch next weight slab -> global_prefetch_b8
      __builtin_prefetch(&Wt[(WMODE == 0) ? (size_t)(k0 + 32) * N : (size_t)(k0 + 32)], 0, 1);
    if (AMODE == 0 || WMODE == 1) wait_async0();
    __syncthreads();

    // ---- compute: wave covers rows msub*16..+15, cols nhalf*64..+63 ----
    v16h af = frag_a_f32(&Asf[msub * 16 + ln][0], hi);
    for (int t = 0; t < 4; ++t) {
      v16h bf = frag_b_f32(&Bsf[nhalf * 64 + t * 16 + ln][0], hi);
      acc[t] = WMMA(af, bf, acc[t]);
    }
  }
  // epilogue
  for (int t = 0; t < 4; ++t) {
    int col = n0 + nhalf * 64 + t * 16 + ln;
    if (col >= N) continue;
    float bv = bias ? bias[col] : 0.f;
    for (int r = 0; r < 8; ++r) {
      int row = m0 + msub * 16 + hi * 8 + r;
      if (row < M) {
        float v = acc[t][r] + bv;
        if (resid) v += resid[(size_t)row * N + col];
        if (relu) v = v > 0.f ? v : 0.f;
        C[(size_t)row * N + col] = v;
      }
    }
  }
}

// ---------------------------------------------------------------------------
// 3x3 conv (256->128) + ReLU as implicit GEMM. Output layout (b, n, pos).
// im2col gather on VALU (zero padding); weight tile via async DMA.
// ---------------------------------------------------------------------------
__global__ __launch_bounds__(256)
void conv3_wmma_k(const float* __restrict__ X, const float* __restrict__ Wt,
                  const float* __restrict__ bias, float* __restrict__ Y)
{
  __shared__ float Asf[64][32];
  __shared__ float Bsf[128][32];
  const int tid = threadIdx.x, lane = tid & 31, wave = tid >> 5;
  const int gpos0 = blockIdx.x * 64;
  const int b = gpos0 >> 16;
  const int pos0 = gpos0 & (PP - 1);
  const int y0 = pos0 >> 10, x0 = pos0 & (WW - 1);
  const int msub = wave & 3, nhalf = wave >> 2;
  const int hi = lane >> 4, ln = lane & 15;
  const float* Xb = X + (size_t)b * C3D_ * PP;

  v8f acc[4];
  for (int t = 0; t < 4; ++t) for (int r = 0; r < 8; ++r) acc[t][r] = 0.f;

  for (int k0 = 0; k0 < KCONV; k0 += 32) {
    __syncthreads();
    for (int i = tid; i < 64 * 32; i += 256) {
      int m = i >> 5, k = i & 31;
      int kk = k0 + k;
      int cin = kk / 9, tap = kk - cin * 9;
      int dy = tap / 3 - 1, dx = tap - (tap / 3) * 3 - 1;
      int xx = x0 + m + dx, yy = y0 + dy;
      float v = 0.f;
      if ((unsigned)yy < (unsigned)HH && (unsigned)xx < (unsigned)WW)
        v = Xb[((size_t)cin << 16) + (yy << 10) + xx];
      Asf[m][k] = v;
    }
    // weight tile: n-major, K-contiguous -> async DMA (4 x 16B per thread)
    for (int j = 0; j < 4; ++j) {
      int chunk = j * 256 + tid;
      int nl   = chunk >> 3;
      int koff = (chunk & 7) << 2;
      async_copy16(Wt + (size_t)nl * KCONV + k0 + koff, &Bsf[nl][koff]);
    }
    if (k0 + 32 < KCONV) __builtin_prefetch(&Wt[(size_t)(k0 + 32)], 0, 1);
    wait_async0();
    __syncthreads();

    v16h af = frag_a_f32(&Asf[msub * 16 + ln][0], hi);
    for (int t = 0; t < 4; ++t) {
      v16h bf = frag_b_f32(&Bsf[nhalf * 64 + t * 16 + ln][0], hi);
      acc[t] = WMMA(af, bf, acc[t]);
    }
  }
  // store: contiguous 8 positions per lane at channel col
  for (int t = 0; t < 4; ++t) {
    int col = nhalf * 64 + t * 16 + ln;
    float bv = bias[col];
    float* dst = Y + ((size_t)b * DD + col) * PP + pos0 + msub * 16 + hi * 8;
    for (int r = 0; r < 8; ++r) { float v = acc[t][r] + bv; dst[r] = v > 0.f ? v : 0.f; }
  }
}

// ---------------------------------------------------------------------------
// Flash attention, one wave per 16-query tile, 32 keys per step, WMMA tiles.
// Q/K/V/O: (B, L, 128) f32, head slice [head*dh, head*dh+dh).
// ---------------------------------------------------------------------------
__global__ __launch_bounds__(32)
void attn_flash_k(const float* __restrict__ Qm, const float* __restrict__ Km,
                  const float* __restrict__ Vm, float* __restrict__ Om,
                  int Lq, int Lk, int dh, const int* __restrict__ keyvalid, float scale)
{
  __shared__ _Float16 Pl[16][32];
  const int lane = threadIdx.x;
  const int hi = lane >> 4, ln = lane & 15;
  const int q0 = blockIdx.x * 16;
  const int head = blockIdx.y;
  const int b = blockIdx.z;
  const int hoff = head * dh;
  const float* Qb = Qm + (size_t)b * Lq * DD;
  const float* Kb = Km + (size_t)b * Lk * DD;
  const float* Vb = Vm + (size_t)b * Lk * DD;
  const int* kvp = keyvalid ? keyvalid + b * Lk : nullptr;
  const int nkc = (dh + 31) >> 5;  // contraction chunks of 32
  const int noc = dh >> 4;         // output col chunks of 16

  // Q A-fragments (pre-scaled)
  v16h qa[2];
  {
    int qrow = q0 + ln; if (qrow >= Lq) qrow = Lq - 1;
    const float* qp = Qb + (size_t)qrow * DD + hoff;
    for (int c = 0; c < 2; ++c) {
      v16h a;
      for (int e = 0; e < 16; ++e) {
        int kd = c * 32 + hi * 16 + e;
        a[e] = (_Float16)((kd < dh) ? qp[kd] * scale : 0.f);
      }
      qa[c] = a;
    }
  }

  float rmax[8], rsum[8];
  v8f oacc[4];
  for (int r = 0; r < 8; ++r) { rmax[r] = -3e30f; rsum[r] = 0.f; }
  for (int c = 0; c < 4; ++c) for (int r = 0; r < 8; ++r) oacc[c][r] = 0.f;

  for (int kt = 0; kt < Lk; kt += 32) {
    // logits: two 16x16 tiles (keys kt+t*16+ln)
    v8f lg[2];
    for (int t = 0; t < 2; ++t) {
      int key = kt + t * 16 + ln; int kc = key < Lk ? key : Lk - 1;
      const float* kp = Kb + (size_t)kc * DD + hoff;
      v8f al; for (int r = 0; r < 8; ++r) al[r] = 0.f;
      for (int c = 0; c < nkc; ++c) {
        v16h bf;
        for (int e = 0; e < 16; ++e) {
          int kd = c * 32 + hi * 16 + e;
          bf[e] = (_Float16)((kd < dh) ? kp[kd] : 0.f);
        }
        al = WMMA(qa[c], bf, al);
      }
      lg[t] = al;
    }
    // mask + online softmax
    int k0i = kt + ln, k1i = kt + 16 + ln;
    bool ok0 = (k0i < Lk) && (!kvp || kvp[k0i]);
    bool ok1 = (k1i < Lk) && (!kvp || kvp[k1i]);
    float tmax[8];
    for (int r = 0; r < 8; ++r) {
      float a0 = ok0 ? lg[0][r] : -3e30f;
      float a1 = ok1 ? lg[1][r] : -3e30f;
      lg[0][r] = a0; lg[1][r] = a1;
      tmax[r] = fmaxf(a0, a1);
    }
    for (int off = 1; off < 16; off <<= 1)
      for (int r = 0; r < 8; ++r) tmax[r] = fmaxf(tmax[r], __shfl_xor(tmax[r], off, 32));
    float fcorr[8];
    for (int r = 0; r < 8; ++r) {
      float nm = fmaxf(rmax[r], tmax[r]);
      fcorr[r] = __expf(rmax[r] - nm);
      rmax[r] = nm;
      lg[0][r] = ok0 ? __expf(lg[0][r] - nm) : 0.f;
      lg[1][r] = ok1 ? __expf(lg[1][r] - nm) : 0.f;
    }
    float tsum[8];
    for (int r = 0; r < 8; ++r) tsum[r] = lg[0][r] + lg[1][r];
    for (int off = 1; off < 16; off <<= 1)
      for (int r = 0; r < 8; ++r) tsum[r] += __shfl_xor(tsum[r], off, 32);
    for (int r = 0; r < 8; ++r) rsum[r] = rsum[r] * fcorr[r] + tsum[r];
    for (int c = 0; c < noc; ++c) for (int r = 0; r < 8; ++r) oacc[c][r] *= fcorr[r];

    // transpose P through LDS into A-fragment layout
    __syncthreads();
    for (int r = 0; r < 8; ++r) {
      Pl[hi * 8 + r][ln]      = (_Float16)lg[0][r];
      Pl[hi * 8 + r][16 + ln] = (_Float16)lg[1][r];
    }
    __syncthreads();
    v16h pa = cat16(*(const h8*)&Pl[ln][hi * 16], *(const h8*)&Pl[ln][hi * 16 + 8]);

    // O += P x V
    for (int c = 0; c < noc; ++c) {
      v16h vb;
      for (int e = 0; e < 16; ++e) {
        int key = kt + hi * 16 + e;
        vb[e] = (_Float16)((key < Lk) ? Vb[(size_t)key * DD + hoff + c * 16 + ln] : 0.f);
      }
      oacc[c] = WMMA(pa, vb, oacc[c]);
    }
  }
  // normalize + store
  for (int c = 0; c < noc; ++c) {
    for (int r = 0; r < 8; ++r) {
      int row = q0 + hi * 8 + r;
      if (row < Lq) {
        float d = rsum[r];
        Om[((size_t)b * Lq + row) * DD + hoff + c * 16 + ln] = (d > 0.f) ? oacc[c][r] / d : 0.f;
      }
    }
  }
}

// ---------------------------------------------------------------------------
// LayerNorm over rows of 128 (one wave per row, shuffle reductions)
// ---------------------------------------------------------------------------
__global__ __launch_bounds__(256)
void layernorm_k(const float* __restrict__ x, const float* __restrict__ g,
                 const float* __restrict__ bb, float* __restrict__ y, int rows)
{
  int row = blockIdx.x * 8 + (threadIdx.x >> 5);
  int lane = threadIdx.x & 31;
  if (row >= rows) return;
  const float4 v = ((const float4*)(x + (size_t)row * DD))[lane];
  float s = v.x + v.y + v.z + v.w;
  for (int off = 16; off; off >>= 1) s += __shfl_xor(s, off, 32);
  float mean = s * (1.f / 128.f);
  float dx = v.x - mean, dy = v.y - mean, dz = v.z - mean, dw = v.w - mean;
  float vs = dx * dx + dy * dy + dz * dz + dw * dw;
  for (int off = 16; off; off >>= 1) vs += __shfl_xor(vs, off, 32);
  float rstd = rsqrtf(vs * (1.f / 128.f) + 1e-5f);
  float4 gg = ((const float4*)g)[lane], b4 = ((const float4*)bb)[lane];
  float4 o;
  o.x = dx * rstd * gg.x + b4.x; o.y = dy * rstd * gg.y + b4.y;
  o.z = dz * rstd * gg.z + b4.z; o.w = dw * rstd * gg.w + b4.w;
  ((float4*)(y + (size_t)row * DD))[lane] = o;
}

// y[i] = a[i] + p[i % period]
__global__ void add_bcast_k(float* __restrict__ y, const float* __restrict__ a,
                            const float* __restrict__ p, int period, int total)
{
  int i = blockIdx.x * 256 + threadIdx.x;
  if (i < total) y[i] = a[i] + p[i % period];
}

// sine positional encoding for rgb grid (24x80, D=128)
__global__ void sine_pos_k(float* __restrict__ pos)
{
  int i = blockIdx.x * 256 + threadIdx.x;
  if (i >= PR * DD) return;
  int f = i & 127, hw = i >> 7;
  int yy = hw / 80, xx = hw - yy * 80;
  const float twopi = 6.283185307179586f;
  float coord; int j;
  if (f < 64) { coord = (float)(yy + 1) * twopi / (24.f + 1e-6f); j = f; }
  else        { coord = (float)(xx + 1) * twopi / (80.f + 1e-6f); j = f - 64; }
  float dimt = __powf(10000.f, 2.f * floorf(j * 0.5f) / 64.f);
  float v = coord / dimt;
  pos[i] = (j & 1) ? __cosf(v) : __sinf(v);
}

// wave32 ballot compaction of qmask -> idx (NQ per batch) + vmask
__global__ __launch_bounds__(32)
void build_idx_k(const unsigned char* __restrict__ qm, int* __restrict__ idx,
                 int* __restrict__ vmask)
{
  int b = blockIdx.x, lane = threadIdx.x;
  const unsigned char* q = qm + (size_t)b * PP;
  int* ib = idx + b * NQ_;
  int cnt = 0;
  for (int base = 0; base < PP && cnt < NQ_; base += 32) {
    int p = base + lane;
    bool pred = q[p] != 0;
    unsigned m = (unsigned)__ballot(pred);
    int off = __popc(m & ((1u << lane) - 1u));
    if (pred && cnt + off < NQ_) ib[cnt + off] = p;
    cnt += __popc(m);
  }
  int nvalid = cnt < NQ_ ? cnt : NQ_;
  int fill = nvalid;
  for (int base = 0; base < PP && fill < NQ_; base += 32) {
    int p = base + lane;
    bool pred = q[p] == 0;
    unsigned m = (unsigned)__ballot(pred);
    int off = __popc(m & ((1u << lane) - 1u));
    if (pred && fill + off < NQ_) ib[fill + off] = p;
    fill += __popc(m);
  }
  int* vb = vmask + b * NQ_;
  for (int i = lane; i < NQ_; i += 32) vb[i] = (i < nvalid) ? 1 : 0;
}

__global__ void gather_x_k(const float* __restrict__ lf, const int* __restrict__ idx,
                           const int* __restrict__ vm, float* __restrict__ x)
{
  int i = blockIdx.x * 256 + threadIdx.x;
  if (i >= BB * NQ_ * DD) return;
  int d = i & 127, bq = i >> 7, b = bq / NQ_;
  float v = 0.f;
  if (vm[bq]) v = lf[((size_t)b * DD + d) * PP + idx[bq]];
  x[i] = v;
}

__global__ void scatter_x_k(float* __restrict__ lf, const float* __restrict__ x,
                            const int* __restrict__ idx, const int* __restrict__ vm)
{
  int i = blockIdx.x * 256 + threadIdx.x;
  if (i >= BB * NQ_ * DD) return;
  int d = i & 127, bq = i >> 7, b = bq / NQ_;
  if (vm[bq]) lf[((size_t)b * DD + d) * PP + idx[bq]] = x[i];
}

// 1x1 pred conv (128->20) + channel softmax, one thread per position
__global__ __launch_bounds__(256)
void pred_softmax_k(const float* __restrict__ lf, const float* __restrict__ pw,
                    const float* __restrict__ pb, float* __restrict__ out)
{
  __shared__ float wsm[NCLS_ * DD];
  for (int i = threadIdx.x; i < NCLS_ * DD; i += 256) wsm[i] = pw[i];
  __syncthreads();
  int gid = blockIdx.x * 256 + threadIdx.x;
  if (gid >= BB * PP) return;
  int b = gid >> 16, pos = gid & (PP - 1);
  const float* src = lf + (size_t)b * DD * PP + pos;
  float feat[DD];
#pragma unroll 8
  for (int d = 0; d < DD; ++d) feat[d] = src[(size_t)d << 16];
  float lg[NCLS_]; float mx = -3e30f;
  for (int c = 0; c < NCLS_; ++c) {
    float s = pb[c];
#pragma unroll 8
    for (int d = 0; d < DD; ++d) s += wsm[c * DD + d] * feat[d];
    lg[c] = s; mx = fmaxf(mx, s);
  }
  float ssum = 0.f;
  for (int c = 0; c < NCLS_; ++c) { lg[c] = __expf(lg[c] - mx); ssum += lg[c]; }
  float inv = 1.f / ssum;
  float* dst = out + (size_t)b * NCLS_ * PP + pos;
  for (int c = 0; c < NCLS_; ++c) dst[(size_t)c << 16] = lg[c] * inv;
}

// ---------------------------------------------------------------------------
// Host driver
// ---------------------------------------------------------------------------
struct LPars {
  const float *b1,*b2,*bk,*bo,*bq,*bv,*ln1b,*ln1g,*ln2b,*ln2g,*w1,*w2,*wk,*wo,*wq,*wv;
};
static LPars getLP(void* const* d_in, int base) {
  LPars p;
  p.b1  =(const float*)d_in[base+0];  p.b2  =(const float*)d_in[base+1];
  p.bk  =(const float*)d_in[base+2];  p.bo  =(const float*)d_in[base+3];
  p.bq  =(const float*)d_in[base+4];  p.bv  =(const float*)d_in[base+5];
  p.ln1b=(const float*)d_in[base+6];  p.ln1g=(const float*)d_in[base+7];
  p.ln2b=(const float*)d_in[base+8];  p.ln2g=(const float*)d_in[base+9];
  p.w1  =(const float*)d_in[base+10]; p.w2  =(const float*)d_in[base+11];
  p.wk  =(const float*)d_in[base+12]; p.wo  =(const float*)d_in[base+13];
  p.wq  =(const float*)d_in[base+14]; p.wv  =(const float*)d_in[base+15];
  return p;
}

extern "C" void kernel_launch(void* const* d_in, const int* in_sizes, int n_in,
                              void* d_out, int out_size, void* d_ws, size_t ws_size,
                              hipStream_t stream)
{
  (void)in_sizes; (void)n_in; (void)out_size; (void)ws_size;
  // sorted-key pytree flatten order:
  // 0 lidar | 1..64 cross_layers(4x16) | 65 final_ln_b 66 final_ln_g
  // 67 pred_b 68 pred_w 69 query_pos | 70 red2d_b 71 red2d_w 72 red3d_b 73 red3d_w
  // 74..137 self_layers(4x16) | 138 qmask | 139 rgb_feature_raw
  const float* lidar      = (const float*)d_in[0];
  const float* final_ln_b = (const float*)d_in[65];
  const float* final_ln_g = (const float*)d_in[66];
  const float* pred_b     = (const float*)d_in[67];
  const float* pred_w     = (const float*)d_in[68];
  const float* query_pos  = (const float*)d_in[69];
  const float* red2d_b    = (const float*)d_in[70];
  const float* red2d_w    = (const float*)d_in[71];
  const float* red3d_b    = (const float*)d_in[72];
  const float* red3d_w    = (const float*)d_in[73];
  const unsigned char* qmask = (const unsigned char*)d_in[138];
  const float* rgb_raw    = (const float*)d_in[139];
  LPars crossL[4], selfL[4];
  for (int i = 0; i < 4; ++i) { crossL[i] = getLP(d_in, 1 + i * 16);
                                selfL[i]  = getLP(d_in, 74 + i * 16); }
  float* out = (float*)d_out;

  // workspace carve-up (floats)
  float* ws = (float*)d_ws;
  size_t o = 0;
  float* lf     = ws + o; o += (size_t)BB * DD * PP;   // conv3 output
  float* rgbtok = ws + o; o += (size_t)BB * PR * DD;   // rgb tokens (b,pos,d)
  float* rgbpos = ws + o; o += (size_t)PR * DD;
  float* kvk    = ws + o; o += (size_t)BB * PR * DD;   // rgbtok + pos
  float* x      = ws + o; o += (size_t)BB * NQ_ * DD;
  float* x2     = ws + o; o += (size_t)BB * NQ_ * DD;
  float* qb     = ws + o; o += (size_t)BB * NQ_ * DD;
  float* Qp     = ws + o; o += (size_t)BB * NQ_ * DD;
  float* Kp     = ws + o; o += (size_t)BB * NQ_ * DD;
  float* Vp     = ws + o; o += (size_t)BB * NQ_ * DD;
  float* ao     = ws + o; o += (size_t)BB * NQ_ * DD;
  float* h1     = ws + o; o += (size_t)BB * NQ_ * DFF_;
  int* idx      = (int*)(ws + o); o += BB * NQ_;
  int* vmask    = (int*)(ws + o); o += BB * NQ_;

  const int TQ = BB * NQ_ * DD;       // 768000
  const int M_Q = BB * NQ_;           // 6000
  const int M_R = BB * PR;            // 3840

  // feature reductions
  conv3_wmma_k<<<dim3(BB * PP / 64), 256, 0, stream>>>(lidar, red3d_w, red3d_b, lf);
  gemm_wmma_k<1,1><<<dim3(M_R / 64, 1), 256, 0, stream>>>(rgb_raw, red2d_w, red2d_b,
                                                          nullptr, rgbtok, M_R, DD, 1024, PR, 1);
  sine_pos_k<<<dim3((PR * DD + 255) / 256), 256, 0, stream>>>(rgbpos);
  build_idx_k<<<dim3(BB), 32, 0, stream>>>(qmask, idx, vmask);
  gather_x_k<<<dim3((TQ + 255) / 256), 256, 0, stream>>>(lf, idx, vmask, x);
  add_bcast_k<<<dim3((BB * PR * DD + 255) / 256), 256, 0, stream>>>(kvk, rgbtok, rgbpos,
                                                                    PR * DD, BB * PR * DD);

  const int HEADSv[4] = {2, 4, 4, 8};
  for (int li = 0; li < 8; ++li) {
    bool selfa = li < 4;
    int lidx = selfa ? li : li - 4;
    LPars p = selfa ? selfL[lidx] : crossL[lidx];
    int hh = HEADSv[lidx], dh = DD / hh;
    float scale = 1.0f / sqrtf((float)dh);

    layernorm_k<<<dim3((M_Q + 7) / 8), 256, 0, stream>>>(x, p.ln1g, p.ln1b, x2, M_Q);
    add_bcast_k<<<dim3((TQ + 255) / 256), 256, 0, stream>>>(qb, x2, query_pos, NQ_ * DD, TQ);
    gemm_wmma_k<0,0><<<dim3((M_Q + 63) / 64, 1), 256, 0, stream>>>(qb, p.wq, p.bq, nullptr,
                                                                   Qp, M_Q, DD, DD, 0, 0);
    if (selfa) {
      gemm_wmma_k<0,0><<<dim3((M_Q + 63) / 64, 1), 256, 0, stream>>>(qb, p.wk, p.bk, nullptr,
                                                                     Kp, M_Q, DD, DD, 0, 0);
      gemm_wmma_k<0,0><<<dim3((M_Q + 63) / 64, 1), 256, 0, stream>>>(x2, p.wv, p.bv, nullptr,
                                                                     Vp, M_Q, DD, DD, 0, 0);
      attn_flash_k<<<dim3((NQ_ + 15) / 16, hh, BB), 32, 0, stream>>>(Qp, Kp, Vp, ao,
                                                                     NQ_, NQ_, dh, vmask, scale);
    } else {
      gemm_wmma_k<0,0><<<dim3(M_R / 64, 1), 256, 0, stream>>>(kvk, p.wk, p.bk, nullptr,
                                                              Kp, M_R, DD, DD, 0, 0);
      gemm_wmma_k<0,0><<<dim3(M_R / 64, 1), 256, 0, stream>>>(rgbtok, p.wv, p.bv, nullptr,
                                                              Vp, M_R, DD, DD, 0, 0);
      attn_flash_k<<<dim3((NQ_ + 15) / 16, hh, BB), 32, 0, stream>>>(Qp, Kp, Vp, ao,
                                                                     NQ_, PR, dh, nullptr, scale);
    }
    // x = x + ao @ wo + bo
    gemm_wmma_k<0,0><<<dim3((M_Q + 63) / 64, 1), 256, 0, stream>>>(ao, p.wo, p.bo, x,
                                                                   x, M_Q, DD, DD, 0, 0);
    // FFN
    layernorm_k<<<dim3((M_Q + 7) / 8), 256, 0, stream>>>(x, p.ln2g, p.ln2b, x2, M_Q);
    gemm_wmma_k<0,0><<<dim3((M_Q + 63) / 64, DFF_ / 128), 256, 0, stream>>>(x2, p.w1, p.b1,
                                                                  nullptr, h1, M_Q, DFF_, DD, 0, 1);
    gemm_wmma_k<0,0><<<dim3((M_Q + 63) / 64, 1), 256, 0, stream>>>(h1, p.w2, p.b2, x,
                                                                   x, M_Q, DD, DFF_, 0, 0);
  }

  layernorm_k<<<dim3((M_Q + 7) / 8), 256, 0, stream>>>(x, final_ln_g, final_ln_b, x2, M_Q);
  scatter_x_k<<<dim3((TQ + 255) / 256), 256, 0, stream>>>(lf, x2, idx, vmask);
  pred_softmax_k<<<dim3((BB * PP + 255) / 256), 256, 0, stream>>>(lf, pred_w, pred_b, out);
}
